// Seq2SeqEncoder_49813030699622
// MI455X (gfx1250) — matrix operand order repaired and checked by reading, not compiled, not run on hardware
//
#include <hip/hip_runtime.h>
#include <hip/hip_bf16.h>
#include <cstdint>
#include <cstddef>

// ---------------------------------------------------------------------------
// 2-layer GRU (Keras reset_after, gates z/r/n) for MI455X (gfx1250, wave32).
//   * layer-0 input projection hoisted out of the time loop: one big
//     bf16 WMMA GEMM  gx0 = emb[X] @ Wx0  (M=65536, N=3072, K=1024)
//   * per timestep: 2 fused kernels (layer0, layer1), each wave owns
//     complete 16x16 gate tiles (z,r,n) -> gating + state update in-register
//   * weights pre-packed into per-lane WMMA B-fragment layout (2 x b128/frag)
// ---------------------------------------------------------------------------

typedef __attribute__((ext_vector_type(16))) __bf16 v16bf;
typedef __attribute__((ext_vector_type(8)))  float  v8f;

#define B_   128
#define T_   512
#define E_   1024
#define H_   1024
#define G3   3072            // 3*H
#define KT_  32              // K tiles of 32 (E_ = H_ = 1024)
#define NT_  192             // 16-wide N tiles over 3H

union FragAB {
  v16bf v;
  uint4 q[2];
  unsigned short s[16];
};

__device__ __forceinline__ unsigned short f2bf(float f) {
  unsigned int u = __builtin_bit_cast(unsigned int, f);
  u = (u + 0x7FFFu + ((u >> 16) & 1u)) >> 16;       // round-to-nearest-even
  return (unsigned short)u;
}
__device__ __forceinline__ float bf2f(unsigned short s) {
  unsigned int u = ((unsigned int)s) << 16;
  return __builtin_bit_cast(float, u);
}
__device__ __forceinline__ float sigmoid_f(float x) {
  return 1.0f / (1.0f + __expf(-x));
}
__device__ __forceinline__ float tanh_f(float x) {
  return 1.0f - 2.0f / (__expf(2.0f * x) + 1.0f);
}

// B fragment (32x16 bf16 tile) from pre-packed weights: 32 contiguous
// bytes per lane -> two global b128 loads.
__device__ __forceinline__ FragAB load_b_packed(const unsigned short* __restrict__ P,
                                                int tile, int lane) {
  FragAB f;
  const uint4* p = reinterpret_cast<const uint4*>(P + (size_t)tile * 512 + lane * 16);
  f.q[0] = p[0];
  f.q[1] = p[1];
  return f;
}

// A fragment (16x32 bf16 tile) built from a row-major f32 state matrix
// (stride H_). Lane l holds row l&15; K chunks depend on lane group.
__device__ __forceinline__ FragAB load_a_global_f32(const float* __restrict__ blockbase,
                                                    int lane, int kt) {
  const float* row = blockbase + (size_t)(lane & 15) * H_;
  int c = kt * 32 + ((lane & 16) ? 8 : 0);
  float4 a0 = *reinterpret_cast<const float4*>(row + c);
  float4 a1 = *reinterpret_cast<const float4*>(row + c + 4);
  float4 c0 = *reinterpret_cast<const float4*>(row + c + 16);
  float4 c1 = *reinterpret_cast<const float4*>(row + c + 20);
  FragAB f;
  f.s[0]  = f2bf(a0.x); f.s[1]  = f2bf(a0.y); f.s[2]  = f2bf(a0.z); f.s[3]  = f2bf(a0.w);
  f.s[4]  = f2bf(a1.x); f.s[5]  = f2bf(a1.y); f.s[6]  = f2bf(a1.z); f.s[7]  = f2bf(a1.w);
  f.s[8]  = f2bf(c0.x); f.s[9]  = f2bf(c0.y); f.s[10] = f2bf(c0.z); f.s[11] = f2bf(c0.w);
  f.s[12] = f2bf(c1.x); f.s[13] = f2bf(c1.y); f.s[14] = f2bf(c1.z); f.s[15] = f2bf(c1.w);
  return f;
}

#define WMMA_BF16(acc, a, b) \
  __builtin_amdgcn_wmma_f32_16x16x32_bf16(false, (a).v, false, (b).v, (short)0, (acc), false, false)

// ---------------------------------------------------------------------------
// Kernel: pack f32 [K, N] weight into per-lane WMMA B-fragment layout (bf16).
// Tile (kt, nt): 512 halfwords, idx = (nt*KT + kt)*512 + lane*16 + h,
//   n = nt*16 + (lane&15),  k = kt*32 + (lane>=16 ? 8 : 0) + (h<8 ? h : h+8).
// ---------------------------------------------------------------------------
__global__ __launch_bounds__(256) void pack_w_bf16(const float* __restrict__ W,
                                                   unsigned short* __restrict__ P,
                                                   int K, int N) {
  int idx = blockIdx.x * 256 + threadIdx.x;
  if (idx >= K * N) return;
  int tile = idx >> 9;
  int w    = idx & 511;
  int lane = w >> 4;
  int h    = w & 15;
  int ktn  = K >> 5;
  int nt   = tile / ktn;
  int kt   = tile - nt * ktn;
  int n = nt * 16 + (lane & 15);
  int k = kt * 32 + ((lane & 16) ? 8 : 0) + (h < 8 ? h : h + 8);
  P[idx] = f2bf(W[(size_t)k * N + n]);
}

__global__ __launch_bounds__(256) void zero_f32(float* __restrict__ p, int n) {
  int i = blockIdx.x * 256 + threadIdx.x;
  if (i < n) p[i] = 0.0f;
}

// ---------------------------------------------------------------------------
// Kernel: gx0 = emb[X] @ Wx0   (bf16 WMMA, M = B*T = 65536, N = 3072, K = 1024)
// One block per 16-row M-block: gather+convert 16 embedding rows into LDS,
// then 8 waves sweep the 192 N tiles (24 each), 32 K tiles per output tile.
// ---------------------------------------------------------------------------
__global__ __launch_bounds__(256) void gx0_gemm(const int* __restrict__ X,
                                                const float* __restrict__ emb,
                                                const unsigned short* __restrict__ Wx0p,
                                                unsigned short* __restrict__ gx0) {
  __shared__ int sTok[16];
  __shared__ __align__(16) unsigned short sA[16][E_ + 8];   // padded bf16 A block

  int tid = threadIdx.x;
  int mb  = blockIdx.x;                       // 16-row block of M = B*T
  if (tid < 16) sTok[tid] = X[mb * 16 + tid];
  __syncthreads();
  for (int i = tid; i < 16 * E_; i += 256) {
    int r = i >> 10;
    int c = i & (E_ - 1);
    sA[r][c] = f2bf(emb[(size_t)sTok[r] * E_ + c]);
  }
  __syncthreads();

  int wid = tid >> 5, lane = tid & 31;
  for (int nt = wid; nt < NT_; nt += 8) {
    v8f acc = {};
    for (int kt = 0; kt < KT_; ++kt) {
      FragAB a;
      const unsigned short* row = &sA[lane & 15][0];
      int c = kt * 32 + ((lane & 16) ? 8 : 0);
      a.q[0] = *reinterpret_cast<const uint4*>(row + c);        // ds_load_b128
      a.q[1] = *reinterpret_cast<const uint4*>(row + c + 16);   // ds_load_b128
      FragAB b = load_b_packed(Wx0p, nt * KT_ + kt, lane);
      acc = WMMA_BF16(acc, a, b);
    }
    int col   = nt * 16 + (lane & 15);
    int rbase = mb * 16 + ((lane & 16) ? 8 : 0);
#pragma unroll
    for (int r = 0; r < 8; ++r)
      gx0[(size_t)(rbase + r) * G3 + col] = f2bf(acc[r]);
  }
}

// ---------------------------------------------------------------------------
// Kernel: one GRU step, layer 0 (input gates precomputed in gx0).
// Grid (8,8): blockIdx.x = M-block of 16 batch rows, blockIdx.y * 8 + wave
// = 16-wide column tile of H. Each wave: 3 gate tiles (z,r,n), 96 WMMA,
// then gating + state update from accumulator registers.
// ---------------------------------------------------------------------------
__global__ __launch_bounds__(256) void gru_step_l0(int t,
    const float* __restrict__ h_prev, float* __restrict__ h_next,
    const unsigned short* __restrict__ Whp,
    const unsigned short* __restrict__ gx0,
    const float* __restrict__ bias /* [2][3H] */) {
  int mb  = blockIdx.x;
  int wid = threadIdx.x >> 5, lane = threadIdx.x & 31;
  int nt  = blockIdx.y * 8 + wid;                 // 0..63 column tile of H
  const float* arow = h_prev + (size_t)mb * 16 * H_;

  v8f az = {}, ar = {}, an = {};
  for (int kt = 0; kt < KT_; ++kt) {
    FragAB a  = load_a_global_f32(arow, lane, kt);
    FragAB bz = load_b_packed(Whp, (nt)       * KT_ + kt, lane);
    FragAB br = load_b_packed(Whp, (nt + 64)  * KT_ + kt, lane);
    FragAB bn = load_b_packed(Whp, (nt + 128) * KT_ + kt, lane);
    if (kt + 1 < KT_)
      __builtin_prefetch(Whp + (size_t)(nt * KT_ + kt + 1) * 512, 0, 0);
    az = WMMA_BF16(az, a, bz);
    ar = WMMA_BF16(ar, a, br);
    an = WMMA_BF16(an, a, bn);
  }

  int j = nt * 16 + (lane & 15);
  float bxz = bias[j],            bhz = bias[G3 + j];
  float bxr = bias[H_ + j],       bhr = bias[G3 + H_ + j];
  float bxn = bias[2 * H_ + j],   bhn = bias[G3 + 2 * H_ + j];
  int rbase = mb * 16 + ((lane & 16) ? 8 : 0);
#pragma unroll
  for (int r = 0; r < 8; ++r) {
    int b = rbase + r;
    size_t gbase = ((size_t)b * T_ + t) * G3;
    float gxz = bf2f(gx0[gbase + j]);
    float gxr = bf2f(gx0[gbase + H_ + j]);
    float gxn = bf2f(gx0[gbase + 2 * H_ + j]);
    float z  = sigmoid_f(gxz + bxz + az[r] + bhz);
    float rg = sigmoid_f(gxr + bxr + ar[r] + bhr);
    float n  = tanh_f((gxn + bxn) + rg * (an[r] + bhn));
    float hp = h_prev[(size_t)b * H_ + j];
    h_next[(size_t)b * H_ + j] = z * hp + (1.0f - z) * n;
  }
}

// ---------------------------------------------------------------------------
// Kernel: one GRU step, layer 1 (x = fresh layer-0 state). Each wave owns 6
// gate tiles (gx z/r/n via Wx1, gh z/r/n via Wh1): 192 WMMA, then gating,
// state update, and the [B,T,H] sequence output write.
// ---------------------------------------------------------------------------
__global__ __launch_bounds__(256) void gru_step_l1(int t,
    const float* __restrict__ x_in,           // layer-0 h for this step
    const float* __restrict__ h_prev, float* __restrict__ h_next,
    const unsigned short* __restrict__ Wxp,
    const unsigned short* __restrict__ Whp,
    const float* __restrict__ bias /* [2][3H] */,
    float* __restrict__ out) {
  int mb  = blockIdx.x;
  int wid = threadIdx.x >> 5, lane = threadIdx.x & 31;
  int nt  = blockIdx.y * 8 + wid;
  const float* xrow = x_in  + (size_t)mb * 16 * H_;
  const float* hrow = h_prev + (size_t)mb * 16 * H_;

  v8f xz = {}, xr = {}, xn = {}, hz = {}, hr = {}, hn = {};
  for (int kt = 0; kt < KT_; ++kt) {
    FragAB ax = load_a_global_f32(xrow, lane, kt);
    FragAB ah = load_a_global_f32(hrow, lane, kt);
    FragAB b;
    b = load_b_packed(Wxp, (nt)       * KT_ + kt, lane); xz = WMMA_BF16(xz, ax, b);
    b = load_b_packed(Wxp, (nt + 64)  * KT_ + kt, lane); xr = WMMA_BF16(xr, ax, b);
    b = load_b_packed(Wxp, (nt + 128) * KT_ + kt, lane); xn = WMMA_BF16(xn, ax, b);
    b = load_b_packed(Whp, (nt)       * KT_ + kt, lane); hz = WMMA_BF16(hz, ah, b);
    b = load_b_packed(Whp, (nt + 64)  * KT_ + kt, lane); hr = WMMA_BF16(hr, ah, b);
    b = load_b_packed(Whp, (nt + 128) * KT_ + kt, lane); hn = WMMA_BF16(hn, ah, b);
  }

  int j = nt * 16 + (lane & 15);
  float bxz = bias[j],          bhz = bias[G3 + j];
  float bxr = bias[H_ + j],     bhr = bias[G3 + H_ + j];
  float bxn = bias[2 * H_ + j], bhn = bias[G3 + 2 * H_ + j];
  int rbase = mb * 16 + ((lane & 16) ? 8 : 0);
#pragma unroll
  for (int r = 0; r < 8; ++r) {
    int b = rbase + r;
    float z  = sigmoid_f(xz[r] + bxz + hz[r] + bhz);
    float rg = sigmoid_f(xr[r] + bxr + hr[r] + bhr);
    float n  = tanh_f((xn[r] + bxn) + rg * (hn[r] + bhn));
    float hp = h_prev[(size_t)b * H_ + j];
    float hv = z * hp + (1.0f - z) * n;
    h_next[(size_t)b * H_ + j] = hv;
    out[((size_t)b * T_ + t) * H_ + j] = hv;   // return_sequences output
  }
}

__global__ __launch_bounds__(256) void copy_finals(const float* __restrict__ h0,
                                                   const float* __restrict__ h1,
                                                   float* __restrict__ tail) {
  int i = blockIdx.x * 256 + threadIdx.x;
  if (i < B_ * H_) {
    tail[i]            = h0[i];
    tail[B_ * H_ + i]  = h1[i];
  }
}

// ---------------------------------------------------------------------------
extern "C" void kernel_launch(void* const* d_in, const int* in_sizes, int n_in,
                              void* d_out, int out_size, void* d_ws, size_t ws_size,
                              hipStream_t stream) {
  (void)in_sizes; (void)n_in; (void)out_size; (void)ws_size;

  const int*   X   = (const int*)  d_in[0];
  const float* emb = (const float*)d_in[1];
  const float* Wx0 = (const float*)d_in[2];
  const float* Wh0 = (const float*)d_in[3];
  const float* b0  = (const float*)d_in[4];
  const float* Wx1 = (const float*)d_in[5];
  const float* Wh1 = (const float*)d_in[6];
  const float* b1  = (const float*)d_in[7];
  float* out = (float*)d_out;

  // workspace carve (everything written before read; re-derived every call)
  size_t off = 0;
  auto carve = [&](size_t bytes) -> void* {
    void* p = (char*)d_ws + off;
    off += (bytes + 255) & ~(size_t)255;
    return p;
  };
  const size_t wbytes = (size_t)E_ * G3 * 2;                 // 6.29 MB each
  unsigned short* pWx0 = (unsigned short*)carve(wbytes);
  unsigned short* pWh0 = (unsigned short*)carve(wbytes);
  unsigned short* pWx1 = (unsigned short*)carve(wbytes);
  unsigned short* pWh1 = (unsigned short*)carve(wbytes);
  unsigned short* gx0  = (unsigned short*)carve((size_t)B_ * T_ * G3 * 2);  // 403 MB
  float* h0buf[2] = { (float*)carve((size_t)B_ * H_ * 4),
                      (float*)carve((size_t)B_ * H_ * 4) };
  float* h1buf[2] = { (float*)carve((size_t)B_ * H_ * 4),
                      (float*)carve((size_t)B_ * H_ * 4) };

  // 1) pack weights to bf16 WMMA-B layout (L2-resident thereafter)
  int packBlocks = (E_ * G3 + 255) / 256;
  pack_w_bf16<<<packBlocks, 256, 0, stream>>>(Wx0, pWx0, E_, G3);
  pack_w_bf16<<<packBlocks, 256, 0, stream>>>(Wh0, pWh0, H_, G3);
  pack_w_bf16<<<packBlocks, 256, 0, stream>>>(Wx1, pWx1, H_, G3);
  pack_w_bf16<<<packBlocks, 256, 0, stream>>>(Wh1, pWh1, H_, G3);

  // 2) zero initial states
  int zb = (B_ * H_ + 255) / 256;
  zero_f32<<<zb, 256, 0, stream>>>(h0buf[0], B_ * H_);
  zero_f32<<<zb, 256, 0, stream>>>(h1buf[0], B_ * H_);

  // 3) bulk parallel GEMM: layer-0 input gates for all timesteps
  gx0_gemm<<<(B_ * T_) / 16, 256, 0, stream>>>(X, emb, pWx0, gx0);

  // 4) sequential scan over T (graph-captured; weights + state live in L2)
  int cur = 0;
  for (int t = 0; t < T_; ++t) {
    int nxt = cur ^ 1;
    gru_step_l0<<<dim3(8, 8), 256, 0, stream>>>(t, h0buf[cur], h0buf[nxt],
                                                pWh0, gx0, b0);
    gru_step_l1<<<dim3(8, 8), 256, 0, stream>>>(t, h0buf[nxt], h1buf[cur],
                                                h1buf[nxt], pWx1, pWh1, b1, out);
    cur = nxt;
  }

  // 5) final states (after 512 toggles, cur == 0)
  copy_finals<<<zb, 256, 0, stream>>>(h0buf[cur], h1buf[cur],
                                      out + (size_t)B_ * T_ * H_);
}
